// MemoryLayer_19516331393363
// MI455X (gfx1250) — matrix operand order repaired and checked by reading, not compiled, not run on hardware
//
#include <hip/hip_runtime.h>
#include <hip/hip_bf16.h>
#include <math.h>

typedef __bf16 bf16_t;
typedef __attribute__((ext_vector_type(16))) __bf16 v16bf;
typedef __attribute__((ext_vector_type(8)))  float  v8f;
typedef __attribute__((ext_vector_type(4)))  unsigned int v4u;
typedef __attribute__((ext_vector_type(8)))  int v8i;
typedef __attribute__((ext_vector_type(4)))  int v4i;

#define B_DIM   32
#define H_DIM   512
#define S_DIM   1024
#define O_DIM   1024
#define G_DIM   1536   // 3*H
#define T_STEPS 128
#define HEADS   8
#define P_DIM   8192   // O*HEADS

__device__ __forceinline__ float lrelu(float x) { return x >= 0.f ? x : 0.2f * x; }
__device__ __forceinline__ float sigmoidf(float x) { return 1.f / (1.f + expf(-x)); }

// ---- WMMA helpers (CDNA5 layouts, wave32) -----------------------------------

__device__ __forceinline__ v8f wmma_bf16(v16bf a, v16bf b, v8f c) {
    return __builtin_amdgcn_wmma_f32_16x16x32_bf16(
        /*neg_a=*/false, a, /*neg_b=*/false, b,
        /*c_mod=*/(short)0, c, /*reuse_a=*/false, /*reuse_b=*/false);
}

// A tile 16x32 bf16 from row-major [rows x ld]; lane l<16 holds row l, K chunk
// [k0,k0+16); lanes 16-31 hold K chunk [k0+16,k0+32).
__device__ __forceinline__ v16bf load_a_tile(const bf16_t* src, int row0, int ld, int k0) {
    int lane = threadIdx.x & 31;
    const bf16_t* p = src + (row0 + (lane & 15)) * ld + k0 + ((lane >> 4) << 4);
    v16bf a;
#pragma unroll
    for (int e = 0; e < 8; ++e) a[e] = p[e];
#pragma unroll
    for (int e = 0; e < 8; ++e) a[8 + e] = p[8 + e];
    return a;
}

// B tile 32x16 where B[k][n] = W[n][k], W row-major [N x ld].
// lane l: n = n0 + l%16, contiguous K chunk of 16 starting at k0 + (l/16)*16.
__device__ __forceinline__ v16bf load_b_tileT(const bf16_t* W, int n0, int ld, int k0) {
    int lane = threadIdx.x & 31;
    const bf16_t* p = W + (n0 + (lane & 15)) * ld + k0 + ((lane >> 4) << 4);
    v16bf b;
#pragma unroll
    for (int e = 0; e < 16; ++e) b[e] = p[e];
    return b;
}

// B tile from an LDS panel laid out [128 rows][32 K] contiguous (TDM tile order).
__device__ __forceinline__ v16bf load_b_lds(const bf16_t* Bs, int nLocal0) {
    int lane = threadIdx.x & 31;
    const bf16_t* p = Bs + (nLocal0 + (lane & 15)) * 32 + ((lane >> 4) << 4);
    v16bf b;
#pragma unroll
    for (int e = 0; e < 16; ++e) b[e] = p[e];
    return b;
}

// C/D 16x16 f32: VGPR j -> row row0 + (lane/16)*8 + j, col = col0 + lane%16.
__device__ __forceinline__ void store_c(float* dst, int row0, int ld, int col0, v8f c) {
    int lane = threadIdx.x & 31;
    int col = col0 + (lane & 15);
    int rb = row0 + ((lane >> 4) << 3);
#pragma unroll
    for (int j = 0; j < 8; ++j) dst[(rb + j) * ld + col] = c[j];
}

// ---- Tensor Data Mover: 2-D tile (bf16) global -> LDS ------------------------
// D# per CDNA5 ISA ch.8: group0 = {count|flags, lds_addr, global_addr[56:0], type=2},
// group1 = {data_size=2B, tensor_dim0/1, tile_dim0/1, tensor_dim0_stride}.
// This toolchain exposes the 6-arg builtin: (v4u, v8i, v4i, v4i, v8i, cpol).
__device__ __forceinline__ void tdm_load_2d(unsigned lds_off, const void* gptr,
                                            unsigned tile_d0, unsigned tile_d1,
                                            unsigned tensor_d0, unsigned tensor_d1,
                                            unsigned stride0) {
    unsigned long long ga = (unsigned long long)gptr;
    v4u g0;
    g0[0] = 1u;                                            // count=1, user mode
    g0[1] = lds_off;                                       // lds_addr (bytes)
    g0[2] = (unsigned)(ga & 0xFFFFFFFFull);                // global_addr[31:0]
    g0[3] = (unsigned)((ga >> 32) & 0x01FFFFFFull)         // global_addr[56:32]
          | (2u << 30);                                    // type = 2 ("image")
    v8i g1;
    g1[0] = 0x00010000;                                    // data_size=1 -> 2 bytes
    g1[1] = (int)((tensor_d0 & 0xFFFFu) << 16);            // tensor_dim0[15:0] @63:48
    g1[2] = (int)(((tensor_d0 >> 16) & 0xFFFFu)            // tensor_dim0[31:16]
          | ((tensor_d1 & 0xFFFFu) << 16));                // tensor_dim1[15:0]
    g1[3] = (int)(((tensor_d1 >> 16) & 0xFFFFu)            // tensor_dim1[31:16]
          | (tile_d0 << 16));                              // tile_dim0 @127:112
    g1[4] = (int)(tile_d1 & 0xFFFFu);                      // tile_dim1; tile_dim2=0
    g1[5] = (int)stride0;                                  // tensor_dim0_stride[31:0]
    g1[6] = 0;                                             // stride0 hi / stride1 lo
    g1[7] = 0;
    v4i zz4 = {0, 0, 0, 0};
    v8i zz8 = {0, 0, 0, 0, 0, 0, 0, 0};
    __builtin_amdgcn_tensor_load_to_lds(g0, g1, zz4, zz4, zz8, 0);
}

// ---- misc kernels -----------------------------------------------------------

__global__ void cvt_f32_bf16_kernel(const float* __restrict__ src, bf16_t* __restrict__ dst, int n) {
    int i = blockIdx.x * blockDim.x + threadIdx.x;
    if (i < n) dst[i] = (bf16_t)src[i];
}

// init: z_bn = BN(lrelu(z @ z2h_w^T + b)); h = m = z_bn; prev = EOS. 1 block x 512
__global__ void init_kernel(const float* __restrict__ z, const float* __restrict__ z2h_w,
                            const float* __restrict__ z2h_b,
                            const float* __restrict__ bn1_g, const float* __restrict__ bn1_b,
                            float* __restrict__ z_bn, float* __restrict__ h,
                            float* __restrict__ m, int* __restrict__ prev) {
    int c = threadIdx.x;  // 0..511
    float vals[B_DIM];
    const float* wr = z2h_w + c * 128;
    float bias = z2h_b[c];
    for (int b = 0; b < B_DIM; ++b) {
        const float* zr = z + b * 128;
        float s = bias;
#pragma unroll 4
        for (int k = 0; k < 128; ++k) s += zr[k] * wr[k];
        vals[b] = lrelu(s);
    }
    float mu = 0.f;
    for (int b = 0; b < B_DIM; ++b) mu += vals[b];
    mu *= (1.f / B_DIM);
    float var = 0.f;
    for (int b = 0; b < B_DIM; ++b) { float d = vals[b] - mu; var += d * d; }
    var *= (1.f / B_DIM);
    float inv = rsqrtf(var + 1e-5f);
    float g = bn1_g[c], bb = bn1_b[c];
    for (int b = 0; b < B_DIM; ++b) {
        float v = (vals[b] - mu) * inv * g + bb;
        z_bn[b * H_DIM + c] = v;
        h[b * H_DIM + c] = v;
        m[b * H_DIM + c] = v;
    }
    if (c < B_DIM) prev[c] = O_DIM - 1;  // EOS
}

// per step: convert h,m to bf16 and build broadcast memory row sm (bf16). 256x256
__global__ void step_cvt_kernel(const float* __restrict__ h, const float* __restrict__ m,
                                const float* __restrict__ memory, const int* __restrict__ perm,
                                int t, bf16_t* __restrict__ h_bf, bf16_t* __restrict__ m_bf,
                                bf16_t* __restrict__ sm_bf) {
    int i = blockIdx.x * blockDim.x + threadIdx.x;  // 0..65535
    if (i < B_DIM * H_DIM) {
        h_bf[i] = (bf16_t)h[i];
    } else if (i < 2 * B_DIM * H_DIM) {
        int j = i - B_DIM * H_DIM;
        m_bf[j] = (bf16_t)m[j];
    } else {
        int j = i - 2 * B_DIM * H_DIM;     // 0..32767 -> (b, c)
        int c = j & (S_DIM - 1);
        sm_bf[j] = (bf16_t)memory[perm[t] * S_DIM + c];
    }
}

// per step: si = BN(concat(lrelu(emb[prev]), z_bn)) -> bf16. 4 blocks x 256
__global__ void step_prep_kernel(const float* __restrict__ emb, const int* __restrict__ prev,
                                 const float* __restrict__ z_bn,
                                 const float* __restrict__ bn2_g, const float* __restrict__ bn2_b,
                                 bf16_t* __restrict__ si_bf) {
    int c = blockIdx.x * blockDim.x + threadIdx.x;  // 0..1023
    float vals[B_DIM];
    for (int b = 0; b < B_DIM; ++b) {
        float v;
        if (c < H_DIM) v = lrelu(emb[prev[b] * H_DIM + c]);
        else           v = z_bn[b * H_DIM + (c - H_DIM)];
        vals[b] = v;
    }
    float mu = 0.f;
    for (int b = 0; b < B_DIM; ++b) mu += vals[b];
    mu *= (1.f / B_DIM);
    float var = 0.f;
    for (int b = 0; b < B_DIM; ++b) { float d = vals[b] - mu; var += d * d; }
    var *= (1.f / B_DIM);
    float inv = rsqrtf(var + 1e-5f);
    float g = bn2_g[c], bb = bn2_b[c];
    for (int b = 0; b < B_DIM; ++b)
        si_bf[b * S_DIM + c] = (bf16_t)((vals[b] - mu) * inv * g + bb);
}

// GRU gate GEMMs: gi = X @ w_ih^T (K=1024), gh = Hprev @ w_hh^T (K=512).
// grid (96, 2, 2): ntile, mtile, path. blockDim 64 (wave0 -> gi, wave1 -> gh).
__global__ void gates_kernel(const bf16_t* __restrict__ si_bf, const bf16_t* __restrict__ sm_bf,
                             const bf16_t* __restrict__ h_bf, const bf16_t* __restrict__ m_bf,
                             const bf16_t* __restrict__ w_ih_bf, const bf16_t* __restrict__ w_hh_bf,
                             float* __restrict__ gi, float* __restrict__ gh) {
    int ntile = blockIdx.x, mtile = blockIdx.y, path = blockIdx.z;
    int wave = threadIdx.x >> 5;
    int n0 = ntile * 16, row0 = mtile * 16;
    v8f c = {};
    if (wave == 0) {
        const bf16_t* A = path ? sm_bf : si_bf;
        for (int k0 = 0; k0 < S_DIM; k0 += 32) {
            v16bf a = load_a_tile(A, row0, S_DIM, k0);
            v16bf b = load_b_tileT(w_ih_bf, n0, S_DIM, k0);
            c = wmma_bf16(a, b, c);
        }
        store_c(gi + path * B_DIM * G_DIM, row0, G_DIM, n0, c);
    } else {
        const bf16_t* A = path ? m_bf : h_bf;
        for (int k0 = 0; k0 < H_DIM; k0 += 32) {
            v16bf a = load_a_tile(A, row0, H_DIM, k0);
            v16bf b = load_b_tileT(w_hh_bf, n0, H_DIM, k0);
            c = wmma_bf16(a, b, c);
        }
        store_c(gh + path * B_DIM * G_DIM, row0, G_DIM, n0, c);
    }
}

// GRU elementwise combine (both paths). 128 blocks x 256 = 32768
__global__ void gru_combine_kernel(const float* __restrict__ gi, const float* __restrict__ gh,
                                   const float* __restrict__ b_ih, const float* __restrict__ b_hh,
                                   float* __restrict__ h, float* __restrict__ m) {
    int i = blockIdx.x * blockDim.x + threadIdx.x;  // path(1)|b(5)|j(9)
    int path = i >> 14, b = (i >> 9) & 31, j = i & (H_DIM - 1);
    const float* GI = gi + path * B_DIM * G_DIM + b * G_DIM;
    const float* GH = gh + path * B_DIM * G_DIM + b * G_DIM;
    float r = sigmoidf(GI[j] + b_ih[j] + GH[j] + b_hh[j]);
    float z = sigmoidf(GI[H_DIM + j] + b_ih[H_DIM + j] + GH[H_DIM + j] + b_hh[H_DIM + j]);
    float n = tanhf(GI[2 * H_DIM + j] + b_ih[2 * H_DIM + j] +
                    r * (GH[2 * H_DIM + j] + b_hh[2 * H_DIM + j]));
    float* S = path ? m : h;
    float hv = S[b * H_DIM + j];
    S[b * H_DIM + j] = (1.f - z) * n + z * hv;
}

// act = BN(lrelu(state), bn3) -> bf16, both paths. 4 blocks x 256 = 1024
__global__ void bn3_kernel(const float* __restrict__ h, const float* __restrict__ m,
                           const float* __restrict__ bn3_g, const float* __restrict__ bn3_b,
                           bf16_t* __restrict__ act_bf) {
    int i = blockIdx.x * blockDim.x + threadIdx.x;  // 0..1023
    int path = i >> 9, c = i & (H_DIM - 1);
    const float* S = path ? m : h;
    float vals[B_DIM];
    for (int b = 0; b < B_DIM; ++b) vals[b] = lrelu(S[b * H_DIM + c]);
    float mu = 0.f;
    for (int b = 0; b < B_DIM; ++b) mu += vals[b];
    mu *= (1.f / B_DIM);
    float var = 0.f;
    for (int b = 0; b < B_DIM; ++b) { float d = vals[b] - mu; var += d * d; }
    var *= (1.f / B_DIM);
    float inv = rsqrtf(var + 1e-5f);
    float g = bn3_g[c], bb = bn3_b[c];
    for (int b = 0; b < B_DIM; ++b)
        act_bf[path * B_DIM * H_DIM + b * H_DIM + c] =
            (bf16_t)((vals[b] - mu) * inv * g + bb);
}

// logits = act @ h2o_w^T + h2o_b. grid (64, 2, 2), blockDim 32 (one wave).
__global__ void logits_kernel(const bf16_t* __restrict__ act_bf, const bf16_t* __restrict__ h2o_w_bf,
                              const float* __restrict__ h2o_b, float* __restrict__ logits) {
    int ntile = blockIdx.x, mtile = blockIdx.y, path = blockIdx.z;
    int n0 = ntile * 16, row0 = mtile * 16;
    const bf16_t* A = act_bf + path * B_DIM * H_DIM;
    v8f c = {};
    for (int k0 = 0; k0 < H_DIM; k0 += 32) {
        v16bf a = load_a_tile(A, row0, H_DIM, k0);
        v16bf b = load_b_tileT(h2o_w_bf, n0, H_DIM, k0);
        c = wmma_bf16(a, b, c);
    }
    int lane = threadIdx.x & 31;
    int col = n0 + (lane & 15);
    int rb = row0 + ((lane >> 4) << 3);
    float bias = h2o_b[col];
    float* L = logits + path * B_DIM * O_DIM;
#pragma unroll
    for (int j = 0; j < 8; ++j) L[(rb + j) * O_DIM + col] = c[j] + bias;
}

// softmax + argmax. grid 64 (path|b), blockDim 256. writes bf16 rows + prev.
__global__ void softmax_kernel(const float* __restrict__ logits, int t,
                               bf16_t* __restrict__ outs_bf, bf16_t* __restrict__ mems_bf,
                               int* __restrict__ prev) {
    __shared__ float sf[256];
    __shared__ int   sidx[256];
    int path = blockIdx.x >> 5, b = blockIdx.x & 31;
    const float* L = logits + path * B_DIM * O_DIM + b * O_DIM;
    float mx = -3.4e38f; int ai = 0;
    for (int c = threadIdx.x; c < O_DIM; c += 256) {
        float v = L[c];
        if (v > mx) { mx = v; ai = c; }
    }
    sf[threadIdx.x] = mx; sidx[threadIdx.x] = ai;
    __syncthreads();
    for (int s = 128; s > 0; s >>= 1) {
        if (threadIdx.x < (unsigned)s) {
            if (sf[threadIdx.x + s] > sf[threadIdx.x]) {
                sf[threadIdx.x] = sf[threadIdx.x + s];
                sidx[threadIdx.x] = sidx[threadIdx.x + s];
            }
        }
        __syncthreads();
    }
    float gmax = sf[0]; int garg = sidx[0];
    __syncthreads();
    float acc = 0.f;
    for (int c = threadIdx.x; c < O_DIM; c += 256) acc += expf(L[c] - gmax);
    sf[threadIdx.x] = acc;
    __syncthreads();
    for (int s = 128; s > 0; s >>= 1) {
        if (threadIdx.x < (unsigned)s) sf[threadIdx.x] += sf[threadIdx.x + s];
        __syncthreads();
    }
    float inv = 1.f / sf[0];
    bf16_t* dst = (path ? mems_bf : outs_bf) + (size_t)(t * B_DIM + b) * O_DIM;
    for (int c = threadIdx.x; c < O_DIM; c += 256)
        dst[c] = (bf16_t)(expf(L[c] - gmax) * inv);
    if (path == 0 && threadIdx.x == 0) prev[b] = garg;
}

// Fused projection + cosine similarity, TDM double-buffered B panels.
// Y (4096 x 1024 bf16, rows g = t*32 + b) @ lin_w^T (8192 x 1024) for both outs
// and mems; bias + lrelu into LDS; 8-head cossim -> out[(b*T + t)*O + o].
// grid (256, 64): mtile (16 rows), ntile (128 proj cols = 16 o x 8 heads).
// blockDim 256 = 8 waves; wave w owns proj cols [ntile*128 + w*16, +16).
// Wave 0 drives the Tensor Data Mover: each K-step's 128x32 bf16 B panel (8 KB)
// is DMA'd into LDS (layout = [col][k] contiguous == WMMA B feed layout),
// double buffered, synchronized with s_wait_tensorcnt + block barriers.
__global__ void proj_cossim_kernel(const bf16_t* __restrict__ outs_bf,
                                   const bf16_t* __restrict__ mems_bf,
                                   const bf16_t* __restrict__ lin_w_bf,
                                   const float* __restrict__ lin_b,
                                   float* __restrict__ out) {
    __shared__ bf16_t Bs[2][128 * 32];   // 16 KB double buffer (TDM destination)
    __shared__ float po[16][128];
    __shared__ float pm[16][128];
    int mtile = blockIdx.x, ntile = blockIdx.y;
    int wave = threadIdx.x >> 5, lane = threadIdx.x & 31;
    int row0 = mtile * 16;
    int n0p = ntile * 128;
    bool issuer = (wave == 0);
    unsigned ldsB[2];
    ldsB[0] = (unsigned)(unsigned long long)(void*)&Bs[0][0];
    ldsB[1] = (unsigned)(unsigned long long)(void*)&Bs[1][0];
    const bf16_t* gB = lin_w_bf + (size_t)n0p * O_DIM;

    if (issuer) {
        tdm_load_2d(ldsB[0], gB + 0,  /*tile*/32, 128, /*tensor*/O_DIM, P_DIM, O_DIM);
        tdm_load_2d(ldsB[1], gB + 32, /*tile*/32, 128, /*tensor*/O_DIM, P_DIM, O_DIM);
    }

    v8f co = {}, cm = {};
    for (int i = 0; i < 32; ++i) {
        if (issuer) __builtin_amdgcn_s_wait_tensorcnt(1);  // panel i landed
        __syncthreads();
        v16bf bw = load_b_lds(&Bs[i & 1][0], wave * 16);
        v16bf ao = load_a_tile(outs_bf, row0, O_DIM, i * 32);
        co = wmma_bf16(ao, bw, co);
        v16bf am = load_a_tile(mems_bf, row0, O_DIM, i * 32);
        cm = wmma_bf16(am, bw, cm);
        __syncthreads();                                   // all reads of buf done
        if (issuer && (i + 2) < 32)
            tdm_load_2d(ldsB[i & 1], gB + (i + 2) * 32,
                        32, 128, O_DIM, P_DIM, O_DIM);
    }

    int colL = wave * 16 + (lane & 15);      // 0..127 within tile
    float bias = lin_b[n0p + colL];
    int rb = (lane >> 4) << 3;
#pragma unroll
    for (int j = 0; j < 8; ++j) {
        po[rb + j][colL] = lrelu(co[j] + bias);
        pm[rb + j][colL] = lrelu(cm[j] + bias);
    }
    __syncthreads();
    int r = threadIdx.x >> 4, ol = threadIdx.x & 15;  // 16 rows x 16 o's
    float dot = 0.f, na = 0.f, nb = 0.f;
#pragma unroll
    for (int hh = 0; hh < HEADS; ++hh) {
        float a = po[r][ol * HEADS + hh];
        float bv = pm[r][ol * HEADS + hh];
        dot += a * bv; na += a * a; nb += bv * bv;
    }
    float denom = fmaxf(sqrtf(na) * sqrtf(nb), 1e-8f);
    int g = row0 + r;
    int t = g >> 5, b = g & 31;
    int o = ntile * 16 + ol;
    out[(size_t)(b * T_STEPS + t) * O_DIM + o] = dot / denom;
}

// ---- host launch ------------------------------------------------------------

static inline char* ws_take(char*& p, size_t bytes) {
    char* r = p;
    p += (bytes + 255) & ~(size_t)255;
    return r;
}

extern "C" void kernel_launch(void* const* d_in, const int* in_sizes, int n_in,
                              void* d_out, int out_size, void* d_ws, size_t ws_size,
                              hipStream_t stream) {
    (void)in_sizes; (void)n_in; (void)out_size; (void)ws_size;
    const float* z      = (const float*)d_in[0];
    const float* z2h_w  = (const float*)d_in[1];
    const float* z2h_b  = (const float*)d_in[2];
    const float* bn1_g  = (const float*)d_in[3];
    const float* bn1_b  = (const float*)d_in[4];
    const float* emb    = (const float*)d_in[5];
    const float* bn2_g  = (const float*)d_in[6];
    const float* bn2_b  = (const float*)d_in[7];
    const float* w_ih   = (const float*)d_in[8];
    const float* w_hh   = (const float*)d_in[9];
    const float* b_ih   = (const float*)d_in[10];
    const float* b_hh   = (const float*)d_in[11];
    const float* h2o_w  = (const float*)d_in[12];
    const float* h2o_b  = (const float*)d_in[13];
    const float* bn3_g  = (const float*)d_in[14];
    const float* bn3_b  = (const float*)d_in[15];
    const float* lin_w  = (const float*)d_in[16];
    const float* lin_b  = (const float*)d_in[17];
    const float* memory = (const float*)d_in[18];
    const int*   perm   = (const int*)d_in[19];
    float* outp = (float*)d_out;

    char* p = (char*)d_ws;
    float* z_bn   = (float*)ws_take(p, sizeof(float) * B_DIM * H_DIM);
    float* h      = (float*)ws_take(p, sizeof(float) * B_DIM * H_DIM);
    float* m      = (float*)ws_take(p, sizeof(float) * B_DIM * H_DIM);
    int*   prev   = (int*)  ws_take(p, sizeof(int) * B_DIM);
    float* gi     = (float*)ws_take(p, sizeof(float) * 2 * B_DIM * G_DIM);
    float* gh     = (float*)ws_take(p, sizeof(float) * 2 * B_DIM * G_DIM);
    float* logits = (float*)ws_take(p, sizeof(float) * 2 * B_DIM * O_DIM);
    bf16_t* si_bf   = (bf16_t*)ws_take(p, sizeof(bf16_t) * B_DIM * S_DIM);
    bf16_t* sm_bf   = (bf16_t*)ws_take(p, sizeof(bf16_t) * B_DIM * S_DIM);
    bf16_t* h_bf    = (bf16_t*)ws_take(p, sizeof(bf16_t) * B_DIM * H_DIM);
    bf16_t* m_bf    = (bf16_t*)ws_take(p, sizeof(bf16_t) * B_DIM * H_DIM);
    bf16_t* act_bf  = (bf16_t*)ws_take(p, sizeof(bf16_t) * 2 * B_DIM * H_DIM);
    bf16_t* w_ih_bf = (bf16_t*)ws_take(p, sizeof(bf16_t) * G_DIM * S_DIM);
    bf16_t* w_hh_bf = (bf16_t*)ws_take(p, sizeof(bf16_t) * G_DIM * H_DIM);
    bf16_t* h2o_bf  = (bf16_t*)ws_take(p, sizeof(bf16_t) * O_DIM * H_DIM);
    bf16_t* lin_bf  = (bf16_t*)ws_take(p, sizeof(bf16_t) * P_DIM * O_DIM);
    bf16_t* outs_bf = (bf16_t*)ws_take(p, sizeof(bf16_t) * T_STEPS * B_DIM * O_DIM);
    bf16_t* mems_bf = (bf16_t*)ws_take(p, sizeof(bf16_t) * T_STEPS * B_DIM * O_DIM);

    // weight conversions (once per launch)
    {
        int n;
        n = G_DIM * S_DIM;
        cvt_f32_bf16_kernel<<<(n + 255) / 256, 256, 0, stream>>>(w_ih, w_ih_bf, n);
        n = G_DIM * H_DIM;
        cvt_f32_bf16_kernel<<<(n + 255) / 256, 256, 0, stream>>>(w_hh, w_hh_bf, n);
        n = O_DIM * H_DIM;
        cvt_f32_bf16_kernel<<<(n + 255) / 256, 256, 0, stream>>>(h2o_w, h2o_bf, n);
        n = P_DIM * O_DIM;
        cvt_f32_bf16_kernel<<<(n + 255) / 256, 256, 0, stream>>>(lin_w, lin_bf, n);
    }

    init_kernel<<<1, 512, 0, stream>>>(z, z2h_w, z2h_b, bn1_g, bn1_b, z_bn, h, m, prev);

    for (int t = 0; t < T_STEPS; ++t) {
        step_cvt_kernel<<<256, 256, 0, stream>>>(h, m, memory, perm, t, h_bf, m_bf, sm_bf);
        step_prep_kernel<<<4, 256, 0, stream>>>(emb, prev, z_bn, bn2_g, bn2_b, si_bf);
        gates_kernel<<<dim3(G_DIM / 16, 2, 2), 64, 0, stream>>>(
            si_bf, sm_bf, h_bf, m_bf, w_ih_bf, w_hh_bf, gi, gh);
        gru_combine_kernel<<<128, 256, 0, stream>>>(gi, gh, b_ih, b_hh, h, m);
        bn3_kernel<<<4, 256, 0, stream>>>(h, m, bn3_g, bn3_b, act_bf);
        logits_kernel<<<dim3(O_DIM / 16, 2, 2), 32, 0, stream>>>(act_bf, h2o_bf, h2o_b, logits);
        softmax_kernel<<<64, 256, 0, stream>>>(logits, t, outs_bf, mems_bf, prev);
    }

    proj_cossim_kernel<<<dim3((T_STEPS * B_DIM) / 16, P_DIM / 128), 256, 0, stream>>>(
        outs_bf, mems_bf, lin_bf, lin_b, outp);
}
